// HybridKernelRegression_65481071395701
// MI455X (gfx1250) — matrix operand order repaired and checked by reading, not compiled, not run on hardware
//
#include <hip/hip_runtime.h>
#include <math.h>

#define D_DIM 256
#define M_DIM 8192
#define N_DIM 8192

typedef __attribute__((ext_vector_type(16))) __bf16       v16bf;
typedef __attribute__((ext_vector_type(8)))  float        v8f;
typedef __attribute__((ext_vector_type(4)))  unsigned int v4u;

__device__ __forceinline__ unsigned short f32_to_bf16_rne(float f) {
  unsigned int u = __float_as_uint(f);
  u += 0x7FFFu + ((u >> 16) & 1u);   // round-to-nearest-even
  return (unsigned short)(u >> 16);
}
__device__ __forceinline__ float bf16_to_f32(unsigned short h) {
  return __uint_as_float(((unsigned int)h) << 16);
}

union FragU { v4u u4[2]; v16bf v; };

// A-matrix 16x32 bf16 fragment (ISA 7.12.2): lanes0-15: K = [0..7],[16..23];
// lanes16-31: K = [8..15],[24..31]. Caller bakes the half*8 offset into p.
__device__ __forceinline__ v16bf load_frag_A(const unsigned short* __restrict__ p) {
  FragU f;
  f.u4[0] = *reinterpret_cast<const v4u*>(p);        // 8 bf16
  f.u4[1] = *reinterpret_cast<const v4u*>(p + 16);   // +16 in K
  return f.v;
}
// B-matrix 32x16 bf16 fragment: lanes0-15: K = [0..7],[16..23];
// lanes16-31: K = [8..15],[24..31]. Caller bakes the half*8 offset into p.
__device__ __forceinline__ v16bf load_frag_B(const unsigned short* __restrict__ p) {
  FragU f;
  f.u4[0] = *reinterpret_cast<const v4u*>(p);
  f.u4[1] = *reinterpret_cast<const v4u*>(p + 16);
  return f.v;
}

// One block per row (256 threads = 256 elements): split fp32 -> bf16 hi/lo
// planes and compute exact fp32 squared norms. Rows [0,M) are X, [M,M+N) are Y.
__global__ __launch_bounds__(256) void prep_split_sq(
    const float* __restrict__ X, const float* __restrict__ Y,
    unsigned short* __restrict__ Xhi, unsigned short* __restrict__ Xlo,
    unsigned short* __restrict__ Yhi, unsigned short* __restrict__ Ylo,
    float* __restrict__ xsq, float* __restrict__ ysq)
{
  __shared__ float red[256];
  const int t = threadIdx.x;
  int row = blockIdx.x;
  const float* src;
  unsigned short *hi, *lo;
  float* sq;
  if (row < M_DIM) {
    src = X + (size_t)row * D_DIM;
    hi  = Xhi + (size_t)row * D_DIM;
    lo  = Xlo + (size_t)row * D_DIM;
    sq  = xsq + row;
  } else {
    row -= M_DIM;
    src = Y + (size_t)row * D_DIM;
    hi  = Yhi + (size_t)row * D_DIM;
    lo  = Ylo + (size_t)row * D_DIM;
    sq  = ysq + row;
  }
  const float v = src[t];
  const unsigned short h = f32_to_bf16_rne(v);
  hi[t] = h;
  lo[t] = f32_to_bf16_rne(v - bf16_to_f32(h));
  red[t] = v * v;
  __syncthreads();
  for (int s = 128; s > 0; s >>= 1) {
    if (t < s) red[t] += red[t + s];
    __syncthreads();
  }
  if (t == 0) *sq = red[0];
}

// Fused RBF regression: each block owns 32 output rows and streams all of
// train_X (L2-resident). 8 waves x (32x64) C footprint per N-tile of 512.
// Split-bf16 GEMM (3 WMMAs per K=32 chunk), exp+w epilogue kept in registers.
__global__ __launch_bounds__(256) void rbf_fused(
    const unsigned short* __restrict__ Xhi, const unsigned short* __restrict__ Xlo,
    const unsigned short* __restrict__ Yhi, const unsigned short* __restrict__ Ylo,
    const float* __restrict__ xsq, const float* __restrict__ ysq,
    const float* __restrict__ w, float* __restrict__ out)
{
  const int lane = threadIdx.x & 31;
  const int wave = threadIdx.x >> 5;   // 0..7
  const int lh   = lane >> 4;          // lane half
  const int lr   = lane & 15;
  const int m0   = blockIdx.x * 32;

  // Per-lane ||x||^2 for the 16 M rows this lane's C elements touch:
  // C tile (mi, VGPR r): M = m0 + mi*16 + 8*lh + r
  float xs[16];
  #pragma unroll
  for (int mi = 0; mi < 2; ++mi)
    #pragma unroll
    for (int r = 0; r < 8; ++r)
      xs[mi * 8 + r] = xsq[m0 + mi * 16 + 8 * lh + r];

  float acc[16];
  #pragma unroll
  for (int i = 0; i < 16; ++i) acc[i] = 0.f;

  const size_t aRow0 = (size_t)(m0 + lr) * D_DIM + lh * 8;  // A frag base, mi=0
  const size_t aRow1 = aRow0 + (size_t)16 * D_DIM;          // mi=1

  for (int nt = 0; nt < N_DIM; nt += 512) {
    const int n0 = nt + wave * 64;

    v8f c[2][4];
    #pragma unroll
    for (int mi = 0; mi < 2; ++mi)
      #pragma unroll
      for (int ni = 0; ni < 4; ++ni)
        #pragma unroll
        for (int e = 0; e < 8; ++e) c[mi][ni][e] = 0.f;

    #pragma unroll
    for (int k0 = 0; k0 < D_DIM; k0 += 32) {
      const v16bf ah0 = load_frag_A(Xhi + aRow0 + k0);
      const v16bf al0 = load_frag_A(Xlo + aRow0 + k0);
      const v16bf ah1 = load_frag_A(Xhi + aRow1 + k0);
      const v16bf al1 = load_frag_A(Xlo + aRow1 + k0);
      #pragma unroll
      for (int ni = 0; ni < 4; ++ni) {
        const size_t bOff = (size_t)(n0 + ni * 16 + lr) * D_DIM + k0 + lh * 8;
        const v16bf bh = load_frag_B(Yhi + bOff);
        const v16bf bl = load_frag_B(Ylo + bOff);
        // cross ~= hi*hi + hi*lo + lo*hi (fp32 accumulate); lo*lo ~2^-18 dropped
        c[0][ni] = __builtin_amdgcn_wmma_f32_16x16x32_bf16(false, al0, false, bh, (short)0, c[0][ni], false, false);
        c[0][ni] = __builtin_amdgcn_wmma_f32_16x16x32_bf16(false, ah0, false, bl, (short)0, c[0][ni], false, false);
        c[0][ni] = __builtin_amdgcn_wmma_f32_16x16x32_bf16(false, ah0, false, bh, (short)0, c[0][ni], false, false);
        c[1][ni] = __builtin_amdgcn_wmma_f32_16x16x32_bf16(false, al1, false, bh, (short)0, c[1][ni], false, false);
        c[1][ni] = __builtin_amdgcn_wmma_f32_16x16x32_bf16(false, ah1, false, bl, (short)0, c[1][ni], false, false);
        c[1][ni] = __builtin_amdgcn_wmma_f32_16x16x32_bf16(false, ah1, false, bh, (short)0, c[1][ni], false, false);
      }
    }

    // Epilogue: K = exp(-(xsq + ysq - 2*cross)) (GAMMA=1), acc += K * w[n].
    #pragma unroll
    for (int ni = 0; ni < 4; ++ni) {
      const int n = n0 + ni * 16 + lr;       // this lane's column in C tiles
      const float ysv = ysq[n];
      const float wv  = w[n];
      #pragma unroll
      for (int mi = 0; mi < 2; ++mi)
        #pragma unroll
        for (int r = 0; r < 8; ++r) {
          const float e = __expf(2.0f * c[mi][ni][r] - xs[mi * 8 + r] - ysv);
          acc[mi * 8 + r] = fmaf(e, wv, acc[mi * 8 + r]);
        }
    }
  }

  // Cross-lane (N across lanes) + cross-wave reduction via LDS float atomics.
  __shared__ float sacc[32];
  if (threadIdx.x < 32) sacc[threadIdx.x] = 0.f;
  __syncthreads();
  #pragma unroll
  for (int mi = 0; mi < 2; ++mi)
    #pragma unroll
    for (int r = 0; r < 8; ++r)
      atomicAdd(&sacc[mi * 16 + 8 * lh + r], acc[mi * 8 + r]);
  __syncthreads();
  if (threadIdx.x < 32) out[m0 + threadIdx.x] = sacc[threadIdx.x];
}

extern "C" void kernel_launch(void* const* d_in, const int* in_sizes, int n_in,
                              void* d_out, int out_size, void* d_ws, size_t ws_size,
                              hipStream_t stream) {
  (void)in_sizes; (void)n_in; (void)out_size; (void)ws_size;
  const float* X = (const float*)d_in[0];
  const float* Y = (const float*)d_in[1];
  const float* w = (const float*)d_in[2];
  float* out = (float*)d_out;

  // Workspace layout (16 MB + 64 KB):
  //   Xhi/Xlo/Yhi/Ylo : 4 MB each (bf16 planes), xsq/ysq : 32 KB each (fp32)
  char* ws = (char*)d_ws;
  const size_t E = (size_t)M_DIM * D_DIM;               // 2M elements
  unsigned short* Xhi = (unsigned short*)(ws + 0 * E * 2);
  unsigned short* Xlo = (unsigned short*)(ws + 1 * E * 2);
  unsigned short* Yhi = (unsigned short*)(ws + 2 * E * 2);
  unsigned short* Ylo = (unsigned short*)(ws + 3 * E * 2);
  float* xsq = (float*)(ws + 4 * E * 2);
  float* ysq = (float*)(ws + 4 * E * 2 + (size_t)M_DIM * 4);

  prep_split_sq<<<M_DIM + N_DIM, 256, 0, stream>>>(X, Y, Xhi, Xlo, Yhi, Ylo, xsq, ysq);
  rbf_fused<<<M_DIM / 32, 256, 0, stream>>>(Xhi, Xlo, Yhi, Ylo, xsq, ysq, w, out);
}
